// MambaMixer_43276090475142
// MI455X (gfx1250) — matrix operand order repaired and checked by reading, not compiled, not run on hardware
//
#include <hip/hip_runtime.h>
#include <math.h>

// ---------------- problem dims ----------------
#define Hd 2048
#define Id 4096
#define Nd 16
#define Rd 128
#define Kc 4
#define Ld 8192

typedef __bf16 bf16_t;
typedef __attribute__((ext_vector_type(16))) __bf16 v16bf;
typedef __attribute__((ext_vector_type(4)))  __bf16 v4bf;
typedef __attribute__((ext_vector_type(8)))  float  v8f;

// ---- CDNA5 async global->LDS DMA (no VGPR staging, ASYNCcnt-tracked) ----
__device__ __forceinline__ void async_load_b128(unsigned lds_off, const void* gptr) {
  asm volatile("global_load_async_to_lds_b128 %0, %1, off"
               :: "v"(lds_off), "v"(gptr) : "memory");
}
__device__ __forceinline__ void wait_asynccnt0() {
  asm volatile("s_wait_asynccnt 0x0" ::: "memory");
}

// =====================================================================
// f32 -> bf16 convert, 4 elems/thread (b128 load, b64 store)
// =====================================================================
__global__ void cvt_f32_bf16(const float* __restrict__ src,
                             bf16_t* __restrict__ dst, size_t n4) {
  size_t i = (size_t)blockIdx.x * blockDim.x + threadIdx.x;
  size_t stride = (size_t)gridDim.x * blockDim.x;
  const float4* s4 = (const float4*)src;
  v4bf* d4 = (v4bf*)dst;
  for (; i < n4; i += stride) {
    float4 f = s4[i];
    v4bf o;
    o[0] = (bf16_t)f.x; o[1] = (bf16_t)f.y;
    o[2] = (bf16_t)f.z; o[3] = (bf16_t)f.w;
    d4[i] = o;
  }
}

// =====================================================================
// f32 (K x N) -> bf16 transposed (N x K): one-time weight prep so the
// GEMM hot loop needs no LDS transpose (pure b128 staging both sides).
// =====================================================================
__global__ __launch_bounds__(256)
void cvt_transpose_bf16(const float* __restrict__ src,   // K x N
                        bf16_t* __restrict__ dst,        // N x K
                        int K, int N) {
  __shared__ float tile[32][33];
  const int kb = blockIdx.x * 32, nb = blockIdx.y * 32;
  const int tx = threadIdx.x & 31, ty = threadIdx.x >> 5;   // 32 x 8
#pragma unroll
  for (int j = 0; j < 32; j += 8) {
    int k = kb + ty + j, n = nb + tx;
    tile[ty + j][tx] = (k < K && n < N) ? src[(size_t)k * N + n] : 0.0f;
  }
  __syncthreads();
#pragma unroll
  for (int j = 0; j < 32; j += 8) {
    int n = nb + ty + j, k = kb + tx;
    if (n < N && k < K) dst[(size_t)n * K + k] = (bf16_t)tile[tx][ty + j];
  }
}

// =====================================================================
// bf16 x bf16^T -> f32 GEMM, v_wmma_f32_16x16x32_bf16.
//   C[M,N] = A[M,K] * BT[N,K]^T, row-major (ldc = N).
//   M % 128 == 0, K % 64 == 0; N bounds-checked at issue/store.
//   Optional epilogue: + bias[n], softplus (delta GEMM).
//   256 thr (8 waves); block tile 128x128; wave tile 32x64; BK=64;
//   double-buffered LDS filled by global_load_async_to_lds_b128
//   -> one barrier per K-tile, zero staging VGPRs, no spills.
// =====================================================================
#define BM 128
#define BN 128
#define BK 64
#define LDSS 72   // 64 + 8 pad; row stride 144 B (16B-aligned)

__global__ __launch_bounds__(256)
void gemm_bf16_wmma(const bf16_t* __restrict__ A,    // M x K
                    const bf16_t* __restrict__ BT,   // N x K
                    float* __restrict__ C,
                    int M, int N, int K,
                    const float* __restrict__ bias, int act) {
  __shared__ __align__(16) bf16_t At[2][BM][LDSS];   // [buf][m][k]
  __shared__ __align__(16) bf16_t Bt[2][BN][LDSS];   // [buf][n][k]

  const int tid   = threadIdx.x;
  const int lane  = tid & 31;
  const int wave  = tid >> 5;
  const int waveM = wave & 3;     // 4 waves along M
  const int waveN = wave >> 2;    // 2 waves along N
  const int m0 = blockIdx.x * BM;
  const int n0 = blockIdx.y * BN;
  const int l15   = lane & 15;
  const int lhalf = (lane >= 16) ? 1 : 0;

  // per-thread staging chunk coords (4 chunks of 8 bf16 = 16 B each)
  int crow[4], cko[4];
#pragma unroll
  for (int i = 0; i < 4; ++i) {
    int c = tid + 256 * i;
    crow[i] = c >> 3;
    cko[i]  = (c & 7) * 8;
  }

  v8f acc[2][4];
#pragma unroll
  for (int mi = 0; mi < 2; ++mi)
#pragma unroll
    for (int ni = 0; ni < 4; ++ni)
#pragma unroll
      for (int j = 0; j < 8; ++j) acc[mi][ni][j] = 0.0f;

  const int NK = K / BK;

  // ---- prologue: async-stage tile 0 into buffer 0 ----
#pragma unroll
  for (int i = 0; i < 4; ++i)
    async_load_b128((unsigned)(size_t)&At[0][crow[i]][cko[i]],
                    &A[(size_t)(m0 + crow[i]) * K + cko[i]]);
#pragma unroll
  for (int i = 0; i < 4; ++i)
    if (n0 + crow[i] < N)
      async_load_b128((unsigned)(size_t)&Bt[0][crow[i]][cko[i]],
                      &BT[(size_t)(n0 + crow[i]) * K + cko[i]]);
  wait_asynccnt0();
  __syncthreads();

  for (int kt = 0; kt < NK; ++kt) {
    const int cur = kt & 1;
    const bool hasNext = (kt + 1 < NK);

    if (hasNext) {
      const size_t k0n = (size_t)(kt + 1) * BK;
      // prefetch two tiles ahead toward the WGP caches
      if (kt + 2 < NK) {
        __builtin_prefetch(
            &A[(size_t)(m0 + crow[0]) * K + (size_t)(kt + 2) * BK + cko[0]], 0, 3);
        if (n0 + crow[0] < N)
          __builtin_prefetch(
              &BT[(size_t)(n0 + crow[0]) * K + (size_t)(kt + 2) * BK + cko[0]], 0, 3);
      }
      // async DMA next tile into the other buffer (runs behind the WMMAs)
#pragma unroll
      for (int i = 0; i < 4; ++i)
        async_load_b128((unsigned)(size_t)&At[cur ^ 1][crow[i]][cko[i]],
                        &A[(size_t)(m0 + crow[i]) * K + k0n + cko[i]]);
#pragma unroll
      for (int i = 0; i < 4; ++i)
        if (n0 + crow[i] < N)
          async_load_b128((unsigned)(size_t)&Bt[cur ^ 1][crow[i]][cko[i]],
                          &BT[(size_t)(n0 + crow[i]) * K + k0n + cko[i]]);
    }

    // ---- compute: two 16x16x32 k-substeps from buffer `cur` ----
#pragma unroll
    for (int s = 0; s < 2; ++s) {
      const int kb = s * 32;
      v16bf afrag[2], bfrag[4];
#pragma unroll
      for (int mi = 0; mi < 2; ++mi) {
        int row = waveM * 32 + mi * 16 + l15;
        int ka  = kb + lhalf * 8;            // ISA 7.12.2 A layout
        uint4* d = reinterpret_cast<uint4*>(&afrag[mi]);
        d[0] = *reinterpret_cast<const uint4*>(&At[cur][row][ka]);
        d[1] = *reinterpret_cast<const uint4*>(&At[cur][row][ka + 16]);
      }
#pragma unroll
      for (int ni = 0; ni < 4; ++ni) {
        int col = waveN * 64 + ni * 16 + l15;
        int kc  = kb + lhalf * 8;            // B layout: same K order as A
        uint4* d = reinterpret_cast<uint4*>(&bfrag[ni]);
        d[0] = *reinterpret_cast<const uint4*>(&Bt[cur][col][kc]);
        d[1] = *reinterpret_cast<const uint4*>(&Bt[cur][col][kc + 16]);
      }
#pragma unroll
      for (int mi = 0; mi < 2; ++mi)
#pragma unroll
        for (int ni = 0; ni < 4; ++ni)
          acc[mi][ni] = __builtin_amdgcn_wmma_f32_16x16x32_bf16(
              false, afrag[mi], false, bfrag[ni],
              (short)0, acc[mi][ni], false, false);
    }

    if (hasNext) {
      wait_asynccnt0();     // this wave's DMA into buf[cur^1] done
      __syncthreads();      // all waves done writing + done reading buf[cur]
    }
  }

  // ---- epilogue + store (C/D layout: VGPR v -> M=v, +8 for hi lanes) ----
  for (int mi = 0; mi < 2; ++mi)
    for (int ni = 0; ni < 4; ++ni) {
      int nn = n0 + waveN * 64 + ni * 16 + l15;
      if (nn >= N) continue;
      float bv = bias ? bias[nn] : 0.0f;
      int mb = m0 + waveM * 32 + mi * 16 + lhalf * 8;
#pragma unroll
      for (int v = 0; v < 8; ++v) {
        float x = acc[mi][ni][v] + bv;
        if (act) x = (x > 20.0f) ? x : log1pf(__expf(x));   // softplus
        C[(size_t)(mb + v) * N + nn] = x;
      }
    }
}

// =====================================================================
// Causal depthwise conv (K=4) + SiLU over x half of proj, 4 chans/thread
// =====================================================================
__global__ void conv_silu(const float* __restrict__ proj,
                          const float* __restrict__ cw,
                          const float* __restrict__ cb,
                          float* __restrict__ xc,
                          bf16_t* __restrict__ xcb) {
  size_t n4 = (size_t)Ld * Id / 4;
  size_t idx = (size_t)blockIdx.x * blockDim.x + threadIdx.x;
  size_t stride = (size_t)gridDim.x * blockDim.x;
  for (; idx < n4; idx += stride) {
    int i4 = (int)(idx & (Id / 4 - 1));
    int t  = (int)(idx >> 10);
    int i  = i4 * 4;
    float s[4];
#pragma unroll
    for (int j = 0; j < 4; ++j) s[j] = cb[i + j];
#pragma unroll
    for (int k = 0; k < Kc; ++k) {
      int tt = t - (Kc - 1) + k;
      if (tt >= 0) {
        float4 xv = *reinterpret_cast<const float4*>(
            &proj[(size_t)tt * (2 * Id) + i]);
        s[0] += xv.x * cw[(i + 0) * Kc + k];
        s[1] += xv.y * cw[(i + 1) * Kc + k];
        s[2] += xv.z * cw[(i + 2) * Kc + k];
        s[3] += xv.w * cw[(i + 3) * Kc + k];
      }
    }
    float4 o;
    v4bf ob;
    float* op = &o.x;
#pragma unroll
    for (int j = 0; j < 4; ++j) {
      float r = s[j] * (1.0f / (1.0f + __expf(-s[j])));   // silu
      op[j] = r;
      ob[j] = (bf16_t)r;
    }
    *reinterpret_cast<float4*>(&xc[idx * 4]) = o;
    *reinterpret_cast<v4bf*>(&xcb[idx * 4])  = ob;
  }
}

// =====================================================================
// dt_lr slice (first R cols of ssm, row stride 160) -> bf16 (4/thread)
// =====================================================================
__global__ void extract_dtlr(const float* __restrict__ ssm,
                             bf16_t* __restrict__ dtlr) {
  size_t n4 = (size_t)Ld * Rd / 4;
  size_t idx = (size_t)blockIdx.x * blockDim.x + threadIdx.x;
  size_t stride = (size_t)gridDim.x * blockDim.x;
  for (; idx < n4; idx += stride) {
    int r4 = (int)(idx & (Rd / 4 - 1));
    int t  = (int)(idx >> 5);
    float4 f = *reinterpret_cast<const float4*>(
        &ssm[(size_t)t * (Rd + 2 * Nd) + r4 * 4]);
    v4bf o;
    o[0] = (bf16_t)f.x; o[1] = (bf16_t)f.y;
    o[2] = (bf16_t)f.z; o[3] = (bf16_t)f.w;
    *reinterpret_cast<v4bf*>(&dtlr[idx * 4]) = o;
  }
}

// =====================================================================
// Selective scan: 1 thread = 1 channel; h[16] in registers; B/C rows
// staged through LDS in 64-step chunks (float4 loads).
// =====================================================================
#define TS 64
__global__ __launch_bounds__(64)
void scan_kernel(const float* __restrict__ delta,
                 const float* __restrict__ ssm,
                 const float* __restrict__ xc,
                 const float* __restrict__ A_log,
                 float* __restrict__ y) {
  __shared__ __align__(16) float Bs[TS][Nd];
  __shared__ __align__(16) float Cs[TS][Nd];
  const int tid = threadIdx.x;
  const int ch  = blockIdx.x * 64 + tid;

  float Arow[Nd], h[Nd];
#pragma unroll
  for (int nn = 0; nn < Nd; ++nn) {
    Arow[nn] = -__expf(A_log[(size_t)ch * Nd + nn]);
    h[nn] = 0.0f;
  }

  for (int t0 = 0; t0 < Ld; t0 += TS) {
#pragma unroll
    for (int j = 0; j < 4; ++j) {           // 256 float4 chunks / 64 threads
      int c   = tid + 64 * j;
      int dtt = c >> 2, nn4 = (c & 3) * 4;
      size_t base = (size_t)(t0 + dtt) * (Rd + 2 * Nd);
      *reinterpret_cast<float4*>(&Bs[dtt][nn4]) =
          *reinterpret_cast<const float4*>(&ssm[base + Rd + nn4]);
      *reinterpret_cast<float4*>(&Cs[dtt][nn4]) =
          *reinterpret_cast<const float4*>(&ssm[base + Rd + Nd + nn4]);
    }
    __syncthreads();
    for (int dtt = 0; dtt < TS; ++dtt) {
      int t = t0 + dtt;
      float d  = delta[(size_t)t * Id + ch];
      float u  = xc[(size_t)t * Id + ch];
      float du = d * u;
      float accv = 0.0f;
#pragma unroll
      for (int nn = 0; nn < Nd; ++nn) {
        h[nn] = h[nn] * __expf(d * Arow[nn]) + du * Bs[dtt][nn];
        accv += h[nn] * Cs[dtt][nn];
      }
      y[(size_t)t * Id + ch] = accv;
    }
    __syncthreads();
  }
}

// =====================================================================
// y2 = (y + xc*D) * silu(gate)  -> bf16   (4 elems/thread)
// =====================================================================
__global__ void fuse_gate(const float* __restrict__ y,
                          const float* __restrict__ xc,
                          const float* __restrict__ proj,
                          const float* __restrict__ Dv,
                          bf16_t* __restrict__ y2b) {
  size_t n4 = (size_t)Ld * Id / 4;
  size_t idx = (size_t)blockIdx.x * blockDim.x + threadIdx.x;
  size_t stride = (size_t)gridDim.x * blockDim.x;
  for (; idx < n4; idx += stride) {
    int i4 = (int)(idx & (Id / 4 - 1));
    int t  = (int)(idx >> 10);
    int i  = i4 * 4;
    float4 g  = *reinterpret_cast<const float4*>(
        &proj[(size_t)t * (2 * Id) + Id + i]);
    float4 yv = *reinterpret_cast<const float4*>(&y[idx * 4]);
    float4 xv = *reinterpret_cast<const float4*>(&xc[idx * 4]);
    float4 dv = *reinterpret_cast<const float4*>(&Dv[i]);
    const float* gp = &g.x; const float* yp = &yv.x;
    const float* xp = &xv.x; const float* dp = &dv.x;
    v4bf o;
#pragma unroll
    for (int j = 0; j < 4; ++j) {
      float gg = gp[j];
      float sg = gg * (1.0f / (1.0f + __expf(-gg)));
      o[j] = (bf16_t)((yp[j] + xp[j] * dp[j]) * sg);
    }
    *reinterpret_cast<v4bf*>(&y2b[idx * 4]) = o;
  }
}

// =====================================================================
// host-side launcher
// =====================================================================
extern "C" void kernel_launch(void* const* d_in, const int* in_sizes, int n_in,
                              void* d_out, int out_size, void* d_ws, size_t ws_size,
                              hipStream_t stream) {
  const float* hidden = (const float*)d_in[0];  // (L,H)
  const float* W_in   = (const float*)d_in[1];  // (H,2I)
  const float* conv_w = (const float*)d_in[2];  // (I,K)
  const float* conv_b = (const float*)d_in[3];  // (I,)
  const float* W_x    = (const float*)d_in[4];  // (I,R+2N)
  const float* W_dt   = (const float*)d_in[5];  // (R,I)
  const float* dt_b   = (const float*)d_in[6];  // (I,)
  const float* A_log  = (const float*)d_in[7];  // (I,N)
  const float* Dvec   = (const float*)d_in[8];  // (I,)
  const float* W_out  = (const float*)d_in[9];  // (I,H)
  float* out = (float*)d_out;                   // (L,H)

  // ---- workspace layout (256B aligned) ----
  char* w = (char*)d_ws;
  size_t off = 0;
  auto alloc = [&](size_t bytes) -> void* {
    void* p = w + off;
    off += (bytes + 255) & ~(size_t)255;
    return p;
  };
  bf16_t* Ab    = (bf16_t*)alloc((size_t)Ld * Hd * 2);          //  32 MB
  bf16_t* WinT  = (bf16_t*)alloc((size_t)Hd * 2 * Id * 2);      //  32 MB (2I x H)
  bf16_t* WxT   = (bf16_t*)alloc((size_t)Id * (Rd + 2 * Nd) * 2); // (160 x I)
  bf16_t* WdtT  = (bf16_t*)alloc((size_t)Rd * Id * 2);          // (I x R)
  bf16_t* WoutT = (bf16_t*)alloc((size_t)Id * Hd * 2);          //  16 MB (H x I)
  float*  proj  = (float*)alloc((size_t)Ld * 2 * Id * 4);       // 256 MB
  float*  xc    = (float*)alloc((size_t)Ld * Id * 4);           // 128 MB
  bf16_t* xcb   = (bf16_t*)alloc((size_t)Ld * Id * 2);          //  64 MB
  float*  ssm   = (float*)alloc((size_t)Ld * (Rd + 2 * Nd) * 4);
  bf16_t* dtlr  = (bf16_t*)alloc((size_t)Ld * Rd * 2);
  float*  delta = (float*)alloc((size_t)Ld * Id * 4);           // 128 MB
  float*  ybuf  = (float*)alloc((size_t)Ld * Id * 4);           // 128 MB
  bf16_t* y2b   = (bf16_t*)alloc((size_t)Ld * Id * 2);          //  64 MB
  (void)ws_size; (void)in_sizes; (void)n_in; (void)out_size;

  const int EW_BLOCKS = 1024, EW_THREADS = 256;

  // 1) operand prep: A-side plain convert; B-side transpose+convert
  cvt_f32_bf16<<<EW_BLOCKS, EW_THREADS, 0, stream>>>(hidden, Ab, (size_t)Ld * Hd / 4);
  cvt_transpose_bf16<<<dim3(Hd / 32, (2 * Id) / 32), 256, 0, stream>>>(
      W_in, WinT, Hd, 2 * Id);
  cvt_transpose_bf16<<<dim3(Id / 32, (Rd + 2 * Nd + 31) / 32), 256, 0, stream>>>(
      W_x, WxT, Id, Rd + 2 * Nd);
  cvt_transpose_bf16<<<dim3(Rd / 32, Id / 32), 256, 0, stream>>>(
      W_dt, WdtT, Rd, Id);
  cvt_transpose_bf16<<<dim3(Id / 32, Hd / 32), 256, 0, stream>>>(
      W_out, WoutT, Id, Hd);

  // 2) proj = hidden @ W_in          (8192 x 8192, K=2048)
  gemm_bf16_wmma<<<dim3(Ld / BM, (2 * Id) / BN), 256, 0, stream>>>(
      Ab, WinT, proj, Ld, 2 * Id, Hd, nullptr, 0);

  // 3) x = silu(causal_dw_conv(x))
  conv_silu<<<EW_BLOCKS, EW_THREADS, 0, stream>>>(proj, conv_w, conv_b, xc, xcb);

  // 4) ssm = x @ W_x                 (8192 x 160, K=4096)
  gemm_bf16_wmma<<<dim3(Ld / BM, (Rd + 2 * Nd + BN - 1) / BN), 256, 0, stream>>>(
      xcb, WxT, ssm, Ld, Rd + 2 * Nd, Id, nullptr, 0);

  // 5) delta = softplus(dt_lr @ W_dt + dt_bias)   (epilogue fused in GEMM)
  extract_dtlr<<<512, EW_THREADS, 0, stream>>>(ssm, dtlr);
  gemm_bf16_wmma<<<dim3(Ld / BM, Id / BN), 256, 0, stream>>>(
      dtlr, WdtT, delta, Ld, Id, Rd, dt_b, 1);

  // 6) selective scan (sequential in L, parallel over I channels)
  scan_kernel<<<Id / 64, 64, 0, stream>>>(delta, ssm, xc, A_log, ybuf);

  // 7) y2 = (y + x*D) * silu(gate)
  fuse_gate<<<EW_BLOCKS, EW_THREADS, 0, stream>>>(ybuf, xc, proj, Dvec, y2b);

  // 8) out = y2 @ W_out              (8192 x 2048, K=4096)
  gemm_bf16_wmma<<<dim3(Ld / BM, Hd / BN), 256, 0, stream>>>(
      y2b, WoutT, out, Ld, Hd, Id, nullptr, 0);
}
